// TimeAwareMessageModel_2267742732915
// MI455X (gfx1250) — matrix operand order repaired and checked by reading, not compiled, not run on hardware
//
#include <hip/hip_runtime.h>

// ---------------------------------------------------------------------------
// CDNA5 (gfx1250) fused edge-MLP + scatter for the TimeAwareMessageModel.
//   out[r, 0:128]   += MLP_in (concat(x[col], edge_attr))   for edges row>col
//   out[r, 128:256] += MLP_out(concat(x[col], edge_attr))   for edges row<col
// Strategy: partition edges by mask (halves GEMM work), pre-swizzle weights
// to bf16 WMMA B-fragments, then one fused kernel does
//   gather -> GEMM1(192->256, wmma bf16) -> ReLU -> GEMM2(256->128) -> atomics
// ---------------------------------------------------------------------------

typedef __attribute__((ext_vector_type(16))) __bf16 v16bf;
typedef __attribute__((ext_vector_type(8)))  float  v8f;

#define TILE_E 96      // edges per block (6 M-tiles of 16)
#define BLOCK  384     // 12 waves (wave32)
#define PAIRS  96      // 192 input features / 2 per packed dword

__device__ __forceinline__ unsigned pack_bf16(float a, float b) {
  unsigned ua = __float_as_uint(a);
  unsigned ub = __float_as_uint(b);
  ua += 0x7FFFu + ((ua >> 16) & 1u);       // round-to-nearest-even
  ub += 0x7FFFu + ((ub >> 16) & 1u);
  return (ua >> 16) | (ub & 0xFFFF0000u);
}

__device__ __forceinline__ unsigned short f2bf(float f) {
  unsigned u = __float_as_uint(f);
  u += 0x7FFFu + ((u >> 16) & 1u);
  return (unsigned short)(u >> 16);
}

union FragCast { uint4 u4[2]; v16bf bf; };

// Load one 16x32 (A) or 32x16 (B) bf16 fragment: 8 dwords per lane.
__device__ __forceinline__ v16bf load_frag(const unsigned* p32, int fragIdx, int lane) {
  const uint4* p = (const uint4*)(p32 + (size_t)fragIdx * 256 + lane * 8);
  FragCast fc;
  fc.u4[0] = p[0];
  fc.u4[1] = p[1];
  return fc.bf;
}

__device__ __forceinline__ v8f wmma_bf16(v16bf a, v16bf b, v8f c) {
  return __builtin_amdgcn_wmma_f32_16x16x32_bf16(
      /*neg_a=*/false, a, /*neg_b=*/false, b,
      /*c_mod=*/(short)0, c, /*reuse_a=*/false, /*reuse_b=*/false);
}

// ---------------------------------------------------------------------------
// Kernel 1: partition edges by flow direction.
// ---------------------------------------------------------------------------
__global__ void partition_kernel(const int* __restrict__ ei, int E,
                                 int* __restrict__ counters,
                                 int* __restrict__ list0,   // row < col -> MLP "out"
                                 int* __restrict__ list1) { // row > col -> MLP "in"
  int e = blockIdx.x * blockDim.x + threadIdx.x;
  if (e >= E) return;
  int r = ei[e];
  int c = ei[E + e];
  if (r < c) {
    int p = atomicAdd(counters + 0, 1);
    list0[p] = e;
  } else if (r > c) {
    int p = atomicAdd(counters + 1, 1);
    list1[p] = e;
  }
}

// ---------------------------------------------------------------------------
// Kernel 2: fp32 [K,N] row-major weights -> bf16 WMMA B-fragments.
// Fragment (kt,nt) covers K=kt*32..+31, N=nt*16..+15. Per lane l:
//   n = nt*16 + (l&15); dword v holds K pair kt*32 + (l>>4)*8 + (v&3)*2 + (v>>2)*16.
// ---------------------------------------------------------------------------
__global__ void swizzle_w(const float* __restrict__ src, unsigned* __restrict__ dst,
                          int K, int N) {
  int NT = N >> 4;
  int total = (K >> 5) * NT * 256;
  for (int idx = blockIdx.x * blockDim.x + threadIdx.x; idx < total;
       idx += gridDim.x * blockDim.x) {
    int v    = idx & 7;
    int lane = (idx >> 3) & 31;
    int frag = idx >> 8;
    int kt = frag / NT;
    int nt = frag % NT;
    int k = kt * 32 + (lane >> 4) * 8 + (v & 3) * 2 + (v >> 2) * 16;
    int n = nt * 16 + (lane & 15);
    dst[idx] = pack_bf16(src[(size_t)k * N + n], src[(size_t)(k + 1) * N + n]);
  }
}

// ---------------------------------------------------------------------------
// Kernel 3: fused gather -> MLP (wmma bf16) -> scatter-add.
// ---------------------------------------------------------------------------
__global__ __launch_bounds__(BLOCK)
void edge_mlp_kernel(const float* __restrict__ x,
                     const int*   __restrict__ ei,
                     const float* __restrict__ ea,
                     int E,
                     const int* __restrict__ counters,
                     const int* __restrict__ list0,
                     const int* __restrict__ list1,
                     const unsigned* __restrict__ w1o, const unsigned* __restrict__ w2o,
                     const unsigned* __restrict__ w1i, const unsigned* __restrict__ w2i,
                     const float* __restrict__ b1o, const float* __restrict__ b2o,
                     const float* __restrict__ b1i, const float* __restrict__ b2i,
                     float* __restrict__ out) {
  // 48 KB: first holds A fragments (6mt x 6kt), then reused for H (6mt x 8kt).
  __shared__ unsigned sh[12288];
  __shared__ int shRow[TILE_E];
  __shared__ int shCol[TILE_E];
  __shared__ int shEdge[TILE_E];

  const int listSel   = blockIdx.y;
  const int cnt       = counters[listSel];
  const int tileStart = blockIdx.x * TILE_E;
  if (tileStart >= cnt) return;

  const int*      lst    = listSel ? list1 : list0;
  const unsigned* w1     = listSel ? w1i : w1o;
  const unsigned* w2     = listSel ? w2i : w2o;
  const float*    b1     = listSel ? b1i : b1o;
  const float*    b2     = listSel ? b2i : b2o;
  const int       colOff = listSel ? 0 : 128;  // output = concat(flow_in, flow_out)

  const int tid = threadIdx.x;

  for (int j = tid; j < TILE_E; j += BLOCK) {
    int g = tileStart + j;
    int e = (g < cnt) ? lst[g] : -1;
    shEdge[j] = e;
    shRow[j]  = (e >= 0) ? ei[e] : -1;
    shCol[j]  = (e >= 0) ? ei[E + e] : 0;
  }
  __syncthreads();

  // ---- Stage A: gather x[col] (128f) ++ edge_attr (64f), fp32->bf16,
  //      written directly in WMMA 16-bit A-fragment layout (ISA 7.12.2).
  for (int it = 0; it < (TILE_E * PAIRS) / BLOCK; ++it) {
    int idx = it * BLOCK + tid;
    int m = idx / PAIRS;      // edge slot in tile
    int p = idx % PAIRS;      // K-pair index (k = 2p)
    int e = shEdge[m];
    unsigned d = 0u;
    if (e >= 0) {
      if (p < 64) {
        const float* s = x + (size_t)shCol[m] * 128 + 2 * p;
        d = pack_bf16(s[0], s[1]);
      } else {
        const float* s = ea + (size_t)e * 64 + 2 * (p - 64);
        d = pack_bf16(s[0], s[1]);
      }
    }
    int mt = m >> 4;
    int kt = p >> 4;
    int q  = p & 15;                          // pair index within fragment
    int v    = ((q >= 8) ? 4 : 0) + (q & 3);  // dword within lane
    int half = (q >> 2) & 1;                  // lane half
    int lanew = half * 16 + (m & 15);
    sh[(mt * 6 + kt) * 256 + lanew * 8 + v] = d;
  }
  __syncthreads();

  const int waveId = tid >> 5;
  const int lane   = tid & 31;

  // ---- GEMM1: [96 x 192] @ W1[192 x 256] + b1, then ReLU ----
  const int wm = waveId % 3;   // 2 m-tiles per wave
  const int wn = waveId / 3;   // 4 n-tiles per wave
  v8f acc[2][4];
#pragma unroll
  for (int j = 0; j < 4; ++j) {
    float bb = b1[(wn * 4 + j) * 16 + (lane & 15)];
#pragma unroll
    for (int i = 0; i < 2; ++i)
#pragma unroll
      for (int t = 0; t < 8; ++t) acc[i][j][t] = bb;
  }
#pragma unroll
  for (int kt = 0; kt < 6; ++kt) {
    v16bf a0 = load_frag(sh, (wm * 2 + 0) * 6 + kt, lane);
    v16bf a1 = load_frag(sh, (wm * 2 + 1) * 6 + kt, lane);
#pragma unroll
    for (int j = 0; j < 4; ++j) {
      v16bf bf = load_frag(w1, kt * 16 + (wn * 4 + j), lane);
      acc[0][j] = wmma_bf16(a0, bf, acc[0][j]);
      acc[1][j] = wmma_bf16(a1, bf, acc[1][j]);
    }
  }
  __syncthreads();   // all waves done reading A region

  // ---- ReLU + repack hidden activations as bf16 A-fragments (reuse sh) ----
  unsigned short* shH16 = (unsigned short*)sh;
#pragma unroll
  for (int i = 0; i < 2; ++i) {
#pragma unroll
    for (int j = 0; j < 4; ++j) {
#pragma unroll
      for (int v = 0; v < 8; ++v) {
        float f = fmaxf(acc[i][j][v], 0.0f);
        int m  = (wm * 2 + i) * 16 + v + 8 * (lane >> 4);  // edge row (D layout)
        int k2 = (wn * 4 + j) * 16 + (lane & 15);          // hidden index
        int mt2 = m >> 4;
        int kt2 = k2 >> 5;
        int kk  = k2 & 31;
        int vv  = ((kk >= 16) ? 4 : 0) + ((kk & 7) >> 1);
        int hb  = (kk >> 3) & 1;
        int lane2 = hb * 16 + (m & 15);
        shH16[(((mt2 * 8 + kt2) * 256) + lane2 * 8 + vv) * 2 + (kk & 1)] = f2bf(f);
      }
    }
  }
  __syncthreads();

  // ---- GEMM2: [96 x 256] @ W2[256 x 128] + b2 ----
  const int wm2 = waveId % 6;  // 1 m-tile per wave
  const int wn2 = waveId / 6;  // 4 n-tiles per wave
  v8f acc2[4];
#pragma unroll
  for (int j = 0; j < 4; ++j) {
    float bb = b2[(wn2 * 4 + j) * 16 + (lane & 15)];
#pragma unroll
    for (int t = 0; t < 8; ++t) acc2[j][t] = bb;
  }
#pragma unroll
  for (int kt = 0; kt < 8; ++kt) {
    v16bf a = load_frag(sh, wm2 * 8 + kt, lane);
#pragma unroll
    for (int j = 0; j < 4; ++j) {
      v16bf bf = load_frag(w2, kt * 8 + (wn2 * 4 + j), lane);
      acc2[j] = wmma_bf16(a, bf, acc2[j]);
    }
  }

  // ---- Scatter-add by destination row (segment_sum) ----
#pragma unroll
  for (int j = 0; j < 4; ++j) {
#pragma unroll
    for (int v = 0; v < 8; ++v) {
      int m = wm2 * 16 + v + 8 * (lane >> 4);
      int r = shRow[m];
      if (r >= 0) {
        int n = (wn2 * 4 + j) * 16 + (lane & 15);
        atomicAdd(out + (size_t)r * 256 + colOff + n, acc2[j][v]);
      }
    }
  }
}

// ---------------------------------------------------------------------------
// Host launcher (graph-capture safe: only async ops on `stream`).
// ---------------------------------------------------------------------------
extern "C" void kernel_launch(void* const* d_in, const int* in_sizes, int n_in,
                              void* d_out, int out_size, void* d_ws, size_t ws_size,
                              hipStream_t stream) {
  const float* x   = (const float*)d_in[0];
  const int*   ei  = (const int*)  d_in[1];
  const float* ea  = (const float*)d_in[2];
  const float* W1o = (const float*)d_in[3];
  const float* b1o = (const float*)d_in[4];
  const float* W2o = (const float*)d_in[5];
  const float* b2o = (const float*)d_in[6];
  const float* W1i = (const float*)d_in[7];
  const float* b1i = (const float*)d_in[8];
  const float* W2i = (const float*)d_in[9];
  const float* b2i = (const float*)d_in[10];
  (void)n_in;

  const int E = in_sizes[1] / 2;      // edge_index is [2, E]
  float* out = (float*)d_out;

  // Workspace layout: [counters 16B][list0 E][list1 E][bf16 weight fragments]
  char* base = (char*)d_ws;
  int* counters = (int*)base;
  int* list0 = (int*)(base + 16);
  int* list1 = list0 + E;
  size_t offW = (16 + 8ull * (size_t)E + 255ull) & ~(size_t)255ull;
  unsigned* w1o = (unsigned*)(base + offW);     // 6*16*256 dwords
  unsigned* w2o = w1o + 6 * 16 * 256;           // 8*8*256 dwords
  unsigned* w1i = w2o + 8 * 8 * 256;
  unsigned* w2i = w1i + 6 * 16 * 256;
  (void)ws_size;

  hipMemsetAsync(counters, 0, 16, stream);
  hipMemsetAsync(out, 0, (size_t)out_size * sizeof(float), stream);

  swizzle_w<<<96, 256, 0, stream>>>(W1o, w1o, 192, 256);
  swizzle_w<<<64, 256, 0, stream>>>(W2o, w2o, 256, 128);
  swizzle_w<<<96, 256, 0, stream>>>(W1i, w1i, 192, 256);
  swizzle_w<<<64, 256, 0, stream>>>(W2i, w2i, 256, 128);

  partition_kernel<<<(E + 255) / 256, 256, 0, stream>>>(ei, E, counters, list0, list1);

  dim3 grid((E + TILE_E - 1) / TILE_E, 2);
  edge_mlp_kernel<<<grid, BLOCK, 0, stream>>>(
      x, ei, ea, E, counters, list0, list1,
      w1o, w2o, w1i, w2i, b1o, b2o, b1i, b2i, out);
}
